// AdvancedMuonAttention_39204461477918
// MI455X (gfx1250) — matrix operand order repaired and checked by reading, not compiled, not run on hardware
//
#include <hip/hip_runtime.h>
#include <hip/hip_bf16.h>

typedef __attribute__((ext_vector_type(16))) _Float16 v16h;
typedef __attribute__((ext_vector_type(8)))  float    v8f;

#define BATCH  4
#define SEQ    2048
#define DMODEL 1024
#define NHEAD  16
#define DHEAD  64

// ---- LDS fragment gathers matching CDNA5 WMMA 16-bit layouts -------------
// A-matrix 16x32 f16: lanes 0-15 -> M=lane, K={0..7,16..23}; lanes 16-31 -> K={8..15,24..31}
static __device__ __forceinline__ v16h lds_load_a16(const _Float16* p, int ld, int lane) {
  const int r  = lane & 15;
  const int kh = (lane >> 4) << 3;
  const _Float16* row = p + r * ld;
  v16h a;
#pragma unroll
  for (int e = 0; e < 8; ++e) {
    a[e]     = row[kh + e];
    a[e + 8] = row[16 + kh + e];
  }
  return a;
}

// B-matrix 32x16 f16 gathered from a row-major [n][k] tile:
// lanes 0-15 -> N=lane, K={0..7,16..23} ; lanes 16-31 -> N=lane-16, K={8..15,24..31}
static __device__ __forceinline__ v16h lds_load_b16(const _Float16* p, int ld, int lane) {
  const int c  = lane & 15;
  const int kh = (lane >> 4) << 3;
  const _Float16* row = p + c * ld;
  v16h b;
#pragma unroll
  for (int e = 0; e < 8; ++e) { b[e] = row[kh + e]; b[e + 8] = row[16 + kh + e]; }
  return b;
}

// ---- Generic WMMA GEMM: C[M,N] = A[M,K] @ W[N,K]^T + bias ----------------
// MODE 0: bias only.  MODE 1: tanh(acc+bias)*scale (scale = 0.125/temp[z] if temp)
template<int BM, int BN, int WM, int WN, int MODE>
__global__ __launch_bounds__(WM * WN * 32)
void gemm_wmma_kernel(const float* A, int lda, long a_zoff,
                      const float* __restrict__ Bw, int ldb,
                      const float* __restrict__ bias,
                      float* C, int ldc, long c_zoff,
                      int K, const float* __restrict__ temp) {
  constexpr int BK  = 32;
  constexpr int LDH = BK + 8;                  // padded LDS row (halves)
  constexpr int MT  = (BM / WM) / 16;
  constexpr int NT  = (BN / WN) / 16;
  constexpr int NTHREADS = WM * WN * 32;

  __shared__ _Float16 As[BM * LDH];
  __shared__ _Float16 Bs[BN * LDH];

  const int tid  = threadIdx.x;
  const int lane = tid & 31;
  const int wave = tid >> 5;
  const int wn   = wave % WN;
  const int wm   = wave / WN;
  const int m0   = blockIdx.y * BM;
  const int n0   = blockIdx.x * BN;
  A += (long)blockIdx.z * a_zoff;
  C += (long)blockIdx.z * c_zoff;

  const v8f vzero = {};
  v8f acc[MT][NT];
#pragma unroll
  for (int i = 0; i < MT; ++i)
#pragma unroll
    for (int j = 0; j < NT; ++j) acc[i][j] = vzero;

  for (int k0 = 0; k0 < K; k0 += BK) {
    // stage A tile (f32 -> f16), prefetch next k-tile
#pragma unroll
    for (int i = tid; i < BM * BK / 4; i += NTHREADS) {
      const int r = i >> 3, cv = (i & 7) << 2;
      const float* src = A + (long)(m0 + r) * lda + k0 + cv;
      const float4 f = *(const float4*)src;
      if (k0 + BK < K) __builtin_prefetch((const void*)(src + BK), 0, 3);
      _Float16* d = As + r * LDH + cv;
      d[0] = (_Float16)f.x; d[1] = (_Float16)f.y;
      d[2] = (_Float16)f.z; d[3] = (_Float16)f.w;
    }
    // stage W tile: weights are [n][k] row-major, exactly what lds_load_b16 wants
#pragma unroll
    for (int i = tid; i < BN * BK / 4; i += NTHREADS) {
      const int r = i >> 3, cv = (i & 7) << 2;
      const float* src = Bw + (long)(n0 + r) * ldb + k0 + cv;
      const float4 f = *(const float4*)src;
      if (k0 + BK < K) __builtin_prefetch((const void*)(src + BK), 0, 3);
      _Float16* d = Bs + r * LDH + cv;
      d[0] = (_Float16)f.x; d[1] = (_Float16)f.y;
      d[2] = (_Float16)f.z; d[3] = (_Float16)f.w;
    }
    __syncthreads();

    v16h af[MT], bf[NT];
#pragma unroll
    for (int i = 0; i < MT; ++i)
      af[i] = lds_load_a16(As + (wm * (BM / WM) + i * 16) * LDH, LDH, lane);
#pragma unroll
    for (int j = 0; j < NT; ++j)
      bf[j] = lds_load_b16(Bs + (wn * (BN / WN) + j * 16) * LDH, LDH, lane);

#pragma unroll
    for (int i = 0; i < MT; ++i)
#pragma unroll
      for (int j = 0; j < NT; ++j)
        acc[i][j] = __builtin_amdgcn_wmma_f32_16x16x32_f16(
            false, af[i], false, bf[j], (short)0, acc[i][j], false, false);
    __syncthreads();
  }

  float scale = 1.0f;
  if (MODE == 1 && temp != nullptr) scale = 0.125f / temp[blockIdx.z];
  const int khalf = lane >> 4;
  const int lcol  = lane & 15;
#pragma unroll
  for (int i = 0; i < MT; ++i) {
#pragma unroll
    for (int j = 0; j < NT; ++j) {
#pragma unroll
      for (int r = 0; r < 8; ++r) {
        const int row = m0 + wm * (BM / WM) + i * 16 + r + khalf * 8;
        const int col = n0 + wn * (BN / WN) + j * 16 + lcol;
        float val = acc[i][j][r] + bias[col];
        if (MODE == 1) val = tanhf(val) * scale;
        C[(long)row * ldc + col] = val;
      }
    }
  }
}

// ---- Flash attention with TRANSPOSED score tiles -------------------------
// S^T = K_tile x Q_tile^T  ->  keys land in VGPRs (in-lane), queries in lanes.
// Softmax reductions become in-lane + one xor-16 merge; state is 1 scalar/lane.
// grid: (SEQ/64, BATCH*NHEAD), block: 128 (4 waves, 16 query rows each)
__global__ __launch_bounds__(128)
void flash_attn_kernel(const float* __restrict__ qn, const float* __restrict__ kn,
                       const float* __restrict__ vv, const int* __restrict__ mask,
                       float* __restrict__ ctx) {
  constexpr int LQ  = 72;   // 64 + pad (halves)
  constexpr int LV  = 40;   // 32 + pad
  constexpr int LP  = 40;
  constexpr int LDM = 36;   // 32 + pad (ints)
  __shared__ _Float16 Qs[64 * LQ];                  // [query][dk]
  __shared__ _Float16 Ks[32 * LQ];                  // [key][dk]
  __shared__ _Float16 Vs[64 * LV];                  // [dk][key] (transposed staging)
  __shared__ __align__(16) _Float16 Ps[4][16 * LP]; // per-wave P^T->[query][key]
  __shared__ int Msk[64 * LDM];                     // mask tile [query][key]

  const int tid   = threadIdx.x;
  const int lane  = tid & 31;
  const int wave  = tid >> 5;
  const int b     = blockIdx.y / NHEAD;
  const int h     = blockIdx.y % NHEAD;
  const int q0    = blockIdx.x * 64;
  const int khalf = lane >> 4;
  const int lcol  = lane & 15;
  const size_t base = (size_t)b * SEQ * DMODEL + (size_t)h * DHEAD;
  const int* mbase = mask + ((size_t)b * SEQ + q0) * SEQ;

  // stage Q tile 64x64 once
  for (int i = tid; i < 64 * 16; i += 128) {
    const int r = i >> 4, cv = (i & 15) << 2;
    const float4 f = *(const float4*)(qn + base + (size_t)(q0 + r) * DMODEL + cv);
    _Float16* d = Qs + r * LQ + cv;
    d[0] = (_Float16)f.x; d[1] = (_Float16)f.y;
    d[2] = (_Float16)f.z; d[3] = (_Float16)f.w;
  }
  __syncthreads();
  // Q consumed as B-fragments (N = queries), loaded once
  v16h qb[2];
  qb[0] = lds_load_b16(Qs + wave * 16 * LQ,      LQ, lane);
  qb[1] = lds_load_b16(Qs + wave * 16 * LQ + 32, LQ, lane);

  // per-lane softmax state for query (wave*16 + lcol), replicated in both halves
  float mq = -1e30f, lq = 0.0f;
  const v8f vzero = {};
  v8f acc[4];
#pragma unroll
  for (int n = 0; n < 4; ++n) acc[n] = vzero;

  for (int k0 = 0; k0 < SEQ; k0 += 32) {
    __syncthreads();  // LDS reuse fence vs previous iteration
    // stage K tile [32 keys][64 dk] and V tile transposed [64 dk][32 keys]
    for (int i = tid; i < 32 * 16; i += 128) {
      const int r = i >> 4, cv = (i & 15) << 2;
      const float4 f = *(const float4*)(kn + base + (size_t)(k0 + r) * DMODEL + cv);
      _Float16* d = Ks + r * LQ + cv;
      d[0] = (_Float16)f.x; d[1] = (_Float16)f.y;
      d[2] = (_Float16)f.z; d[3] = (_Float16)f.w;
      const float4 g = *(const float4*)(vv + base + (size_t)(k0 + r) * DMODEL + cv);
      Vs[(cv + 0) * LV + r] = (_Float16)g.x;
      Vs[(cv + 1) * LV + r] = (_Float16)g.y;
      Vs[(cv + 2) * LV + r] = (_Float16)g.z;
      Vs[(cv + 3) * LV + r] = (_Float16)g.w;
    }
    // stage mask tile 64x32 (coalesced int4)
    for (int i = tid; i < 64 * 8; i += 128) {
      const int r = i >> 3, cv = (i & 7) << 2;
      const int4 m4 = *(const int4*)(mbase + (size_t)r * SEQ + k0 + cv);
      int* d = Msk + r * LDM + cv;
      d[0] = m4.x; d[1] = m4.y; d[2] = m4.z; d[3] = m4.w;
    }
    __syncthreads();

    // transposed scores: st[sub] holds (key = sub*16 + r + khalf*8, query = lcol)
    v8f st[2];
#pragma unroll
    for (int sub = 0; sub < 2; ++sub) {
      const v16h ka0 = lds_load_a16(Ks + (sub * 16) * LQ,      LQ, lane);
      const v16h ka1 = lds_load_a16(Ks + (sub * 16) * LQ + 32, LQ, lane);
      st[sub] = vzero;
      st[sub] = __builtin_amdgcn_wmma_f32_16x16x32_f16(false, ka0, false, qb[0], (short)0, st[sub], false, false);
      st[sub] = __builtin_amdgcn_wmma_f32_16x16x32_f16(false, ka1, false, qb[1], (short)0, st[sub], false, false);
    }

    // mask (contiguous per-lane int4 reads) + in-lane max
    float sm[2][8];
    float lmax = -1e30f;
#pragma unroll
    for (int sub = 0; sub < 2; ++sub) {
      const int moff = (wave * 16 + lcol) * LDM + sub * 16 + khalf * 8;
      const int4 mA = *(const int4*)(Msk + moff);
      const int4 mB = *(const int4*)(Msk + moff + 4);
      sm[sub][0] = (mA.x == 0) ? -1e9f : st[sub][0];
      sm[sub][1] = (mA.y == 0) ? -1e9f : st[sub][1];
      sm[sub][2] = (mA.z == 0) ? -1e9f : st[sub][2];
      sm[sub][3] = (mA.w == 0) ? -1e9f : st[sub][3];
      sm[sub][4] = (mB.x == 0) ? -1e9f : st[sub][4];
      sm[sub][5] = (mB.y == 0) ? -1e9f : st[sub][5];
      sm[sub][6] = (mB.z == 0) ? -1e9f : st[sub][6];
      sm[sub][7] = (mB.w == 0) ? -1e9f : st[sub][7];
#pragma unroll
      for (int r = 0; r < 8; ++r) lmax = fmaxf(lmax, sm[sub][r]);
    }
    // merge the two key-halves (lane q holds keys lo, lane q+16 keys hi)
    const float tmax = fmaxf(lmax, __shfl_xor(lmax, 16, 32));
    const float mnew = fmaxf(mq, tmax);
    const float alpha = __expf(mq - mnew);
    mq = mnew;

    float lsum = 0.0f;
#pragma unroll
    for (int sub = 0; sub < 2; ++sub) {
      union { _Float16 hbuf[8]; float4 f4; } pk;
#pragma unroll
      for (int r = 0; r < 8; ++r) {
        const float p = __expf(sm[sub][r] - mnew);
        lsum += p;
        pk.hbuf[r] = (_Float16)p;
      }
      // packed 16B store of 8 consecutive keys for this query row
      *(float4*)(Ps[wave] + lcol * LP + sub * 16 + khalf * 8) = pk.f4;
    }
    lq = lq * alpha + (lsum + __shfl_xor(lsum, 16, 32));

    // redistribute alpha from per-query lanes into accumulator row layout
#pragma unroll
    for (int r = 0; r < 8; ++r) {
      const float av = __shfl(alpha, r + khalf * 8, 32);
#pragma unroll
      for (int n = 0; n < 4; ++n) acc[n][r] *= av;
    }

    // ctx += P(16x32) @ V(32x64): A-frag straight from Ps[query][key]
    const v16h pa = lds_load_a16(Ps[wave], LP, lane);
#pragma unroll
    for (int n = 0; n < 4; ++n) {
      const v16h bv = lds_load_b16(Vs + (n * 16) * LV, LV, lane);
      acc[n] = __builtin_amdgcn_wmma_f32_16x16x32_f16(false, pa, false, bv, (short)0, acc[n], false, false);
    }
  }

  // epilogue: redistribute 1/l into row layout and store ctx
  const float inv = 1.0f / lq;
#pragma unroll
  for (int r = 0; r < 8; ++r) {
    const float iv = __shfl(inv, r + khalf * 8, 32);
    const int qrow = q0 + wave * 16 + r + khalf * 8;
#pragma unroll
    for (int n = 0; n < 4; ++n)
      ctx[base + (size_t)qrow * DMODEL + n * 16 + lcol] = acc[n][r] * iv;
  }
}

// ---- RMSNorm over rows of 1024 -------------------------------------------
__global__ __launch_bounds__(256)
void rmsnorm_kernel(const float* __restrict__ x, const float* __restrict__ w,
                    float* __restrict__ out) {
  __shared__ float red[8];
  const size_t row = blockIdx.x;
  const float* xr = x + row * DMODEL;
  float ss = 0.0f;
  for (int i = threadIdx.x; i < DMODEL; i += 256) { const float t = xr[i]; ss += t * t; }
#pragma unroll
  for (int d = 16; d >= 1; d >>= 1) ss += __shfl_xor(ss, d, 32);
  if ((threadIdx.x & 31) == 0) red[threadIdx.x >> 5] = ss;
  __syncthreads();
  float tot = 0.0f;
#pragma unroll
  for (int i = 0; i < 8; ++i) tot += red[i];
  const float rr = rsqrtf(tot / (float)DMODEL + 1e-8f);
  for (int i = threadIdx.x; i < DMODEL; i += 256) out[row * DMODEL + i] = xr[i] * rr * w[i];
}

extern "C" void kernel_launch(void* const* d_in, const int* in_sizes, int n_in,
                              void* d_out, int out_size, void* d_ws, size_t ws_size,
                              hipStream_t stream) {
  (void)in_sizes; (void)n_in; (void)out_size; (void)ws_size;
  const float* Q    = (const float*)d_in[0];
  const float* K    = (const float*)d_in[1];
  const float* V    = (const float*)d_in[2];
  const int*   mask = (const int*)d_in[3];
  const float* wq   = (const float*)d_in[4];
  const float* bq   = (const float*)d_in[5];
  const float* wk   = (const float*)d_in[6];
  const float* bk   = (const float*)d_in[7];
  const float* wv   = (const float*)d_in[8];
  const float* bv   = (const float*)d_in[9];
  const float* wo   = (const float*)d_in[10];
  const float* bo   = (const float*)d_in[11];
  const float* naqw = (const float*)d_in[12];
  const float* naqb = (const float*)d_in[13];
  const float* nakw = (const float*)d_in[14];
  const float* nakb = (const float*)d_in[15];
  const float* temp = (const float*)d_in[16];
  const float* rmsw = (const float*)d_in[17];

  const size_t NE = (size_t)BATCH * SEQ * DMODEL;   // 8Mi floats = 32 MB
  float* W0 = (float*)d_ws;       // q-proj -> qn (in-place NA)
  float* W1 = W0 + NE;            // k-proj -> kn, later out-proj result
  float* W2 = W1 + NE;            // v-proj
  float* W3 = W2 + NE;            // attention context

  const dim3 gproj(DMODEL / 128, (BATCH * SEQ) / 128, 1);
  gemm_wmma_kernel<128,128,2,4,0><<<gproj, 256, 0, stream>>>(Q, DMODEL, 0, wq, DMODEL, bq, W0, DMODEL, 0, DMODEL, nullptr);
  gemm_wmma_kernel<128,128,2,4,0><<<gproj, 256, 0, stream>>>(K, DMODEL, 0, wk, DMODEL, bk, W1, DMODEL, 0, DMODEL, nullptr);
  gemm_wmma_kernel<128,128,2,4,0><<<gproj, 256, 0, stream>>>(V, DMODEL, 0, wv, DMODEL, bv, W2, DMODEL, 0, DMODEL, nullptr);

  // per-head NeuralAttention: tanh(x @ naw^T + nab); q side also absorbs 1/sqrt(DK)/temp
  const dim3 gna(1, (BATCH * SEQ) / 64, NHEAD);
  gemm_wmma_kernel<64,64,2,2,1><<<gna, 128, 0, stream>>>(W0, DMODEL, DHEAD, naqw, DHEAD, naqb, W0, DMODEL, DHEAD, DHEAD, temp);
  gemm_wmma_kernel<64,64,2,2,1><<<gna, 128, 0, stream>>>(W1, DMODEL, DHEAD, nakw, DHEAD, nakb, W1, DMODEL, DHEAD, DHEAD, nullptr);

  const dim3 gfa(SEQ / 64, BATCH * NHEAD, 1);
  flash_attn_kernel<<<gfa, 128, 0, stream>>>(W0, W1, W2, mask, W3);

  gemm_wmma_kernel<128,128,2,4,0><<<gproj, 256, 0, stream>>>(W3, DMODEL, 0, wo, DMODEL, bo, W1, DMODEL, 0, DMODEL, nullptr);

  rmsnorm_kernel<<<BATCH * SEQ, 256, 0, stream>>>(W1, rmsw, (float*)d_out);
}
